// CollaborativeAttention_79096117723503
// MI455X (gfx1250) — matrix operand order repaired and checked by reading, not compiled, not run on hardware
//
#include <hip/hip_runtime.h>
#include <hip/hip_bf16.h>
#include <math.h>

typedef unsigned short u16;
typedef unsigned long long u64;
typedef __attribute__((ext_vector_type(16))) __bf16 v16bf;
typedef __attribute__((ext_vector_type(8)))  float  v8f;
typedef __attribute__((ext_vector_type(4)))  unsigned int u32x4;
typedef __attribute__((ext_vector_type(8)))  int i32x8;
typedef __attribute__((ext_vector_type(4)))  int i32x4;

#define HH     16
#define DM     1024
#define DH     64
#define SEQ    512
#define FN     2048
#define TN     2048
#define LNEPS  1e-5f

#if __has_builtin(__builtin_amdgcn_tensor_load_to_lds)
#define HAS_TDM 1
#else
#define HAS_TDM 0
#endif
#if __has_include(<hip/amd_detail/amd_gfx1250_TDM.h>)
#define TDM_6ARG 1
#else
#define TDM_6ARG 0
#endif

// ---------- bf16 helpers ----------
__device__ __forceinline__ u16 f2bf(float x) {
  unsigned u = __float_as_uint(x);
  u += 0x7FFFu + ((u >> 16) & 1u);           // round-to-nearest-even
  return (u16)(u >> 16);
}
__device__ __forceinline__ float bf2f(u16 h) {
  return __uint_as_float(((unsigned)h) << 16);
}

union FragBF { v16bf v; uint4 q[2]; };

// A-matrix 16x32 bf16 fragment (ISA 7.12.2): lane r=lane&15 holds row r.
// kh=lane>>4: elems 0..7 -> K = 8*kh + 0..7 ; elems 8..15 -> K = 16 + 8*kh + 0..7
__device__ __forceinline__ v16bf load_frag_a(const u16* base, int ld, int row0, int k0) {
  int lane = threadIdx.x & 31;
  int r = lane & 15, kh = lane >> 4;
  const u16* p = base + (size_t)(row0 + r) * ld + k0 + kh * 8;
  FragBF f;
  f.q[0] = *reinterpret_cast<const uint4*>(p);
  f.q[1] = *reinterpret_cast<const uint4*>(p + 16);
  return f.v;
}

// B-matrix 32x16 bf16 fragment, loaded from B^T row-major [N,K]:
// lane n=lane&15 holds column n; elems 0..7 -> K = 8*kh + 0..7 ; elems 8..15 -> K = 16 + 8*kh + 0..7
__device__ __forceinline__ v16bf load_frag_b(const u16* base, int ld, int n0, int k0) {
  int lane = threadIdx.x & 31;
  int n = lane & 15, kh = lane >> 4;
  const u16* p = base + (size_t)(n0 + n) * ld + k0 + kh * 8;
  FragBF f;
  f.q[0] = *reinterpret_cast<const uint4*>(p);
  f.q[1] = *reinterpret_cast<const uint4*>(p + 16);
  return f.v;
}

__device__ __forceinline__ v8f wmma_bf16(v16bf a, v16bf b, v8f c) {
  return __builtin_amdgcn_wmma_f32_16x16x32_bf16(false, a, false, b, (short)0, c, false, false);
}

__device__ __forceinline__ void wait_tensorcnt0() {
#if __has_builtin(__builtin_amdgcn_s_wait_tensorcnt)
  __builtin_amdgcn_s_wait_tensorcnt(0);
#else
  asm volatile("s_wait_tensorcnt 0" ::: "memory");
#endif
}

// TDM 2D tile load: global (row-major, stride0 elems, 2-byte data) -> LDS.
// Descriptor per ISA 8.3/8.4: group0 {count=1, lds_addr, global_addr, type=2},
// group1 {data_size=2B, tensor dims == tile dims (tile fully in-bounds), stride0}.
__device__ __forceinline__ void tdm_load_2d(unsigned lds_off, u64 gaddr,
                                            unsigned tile_d0, unsigned tile_d1,
                                            unsigned stride0) {
#if HAS_TDM
  u32x4 g0;
  g0[0] = 1u;                                          // count=1 (valid user D#)
  g0[1] = lds_off;                                     // lds_addr (bytes)
  g0[2] = (unsigned)(gaddr & 0xFFFFFFFFull);           // global_addr[31:0]
  g0[3] = (unsigned)((gaddr >> 32) & 0x01FFFFFFull) | (2u << 30);  // [56:32] + type=2
  i32x8 g1;
  g1[0] = (int)(1u << 16);                             // data_size=1 -> 2 bytes
  g1[1] = (int)((tile_d0 & 0xFFFFu) << 16);            // tensor_dim0[15:0] @ bit48
  g1[2] = (int)((tile_d0 >> 16) | ((tile_d1 & 0xFFFFu) << 16)); // td0 hi | td1 lo
  g1[3] = (int)((tile_d1 >> 16) | (tile_d0 << 16));    // td1 hi | tile_dim0
  g1[4] = (int)(tile_d1 & 0xFFFFu);                    // tile_dim1 (tile_dim2=0)
  g1[5] = (int)stride0;                                // tensor_dim0_stride[31:0]
  g1[6] = 0;                                           // stride0 hi | stride1 lo
  g1[7] = 0;
  i32x4 z4 = {0, 0, 0, 0};
#if TDM_6ARG
  i32x8 z8 = {0, 0, 0, 0, 0, 0, 0, 0};
  __builtin_amdgcn_tensor_load_to_lds(g0, g1, z4, z4, z8, 0);
#else
  __builtin_amdgcn_tensor_load_to_lds(g0, g1, z4, z4, 0);
#endif
#else
  (void)lds_off; (void)gaddr; (void)tile_d0; (void)tile_d1; (void)stride0;
#endif
}

// ---------- gather + convert (hidden_states[idx % 512] -> bf16 [+f32]) ----------
__global__ void gather_convert_kernel(const float* __restrict__ hidden,
                                      const int* __restrict__ pos,
                                      u16* __restrict__ out_bf,
                                      float* __restrict__ out_f32, int n_rows) {
  int idx = blockIdx.x * blockDim.x + threadIdx.x;
  if (idx >= n_rows * DM) return;
  int row = idx >> 10, d = idx & (DM - 1);
  int src = pos[row] & (SEQ - 1);            // % 512 (positions are non-negative)
  float v = hidden[(size_t)src * DM + d];
  out_bf[idx] = f2bf(v);
  if (out_f32) out_f32[idx] = v;
}

// ---------- weight transpose + convert: W[K,N] f32 -> Wt[N,K] bf16 ----------
__global__ void transpose_convert_kernel(const float* __restrict__ W,
                                         u16* __restrict__ Wt, int K, int N) {
  int idx = blockIdx.x * blockDim.x + threadIdx.x;
  if (idx >= K * N) return;
  int k = idx / N, n = idx - k * N;
  Wt[(size_t)n * K + k] = f2bf(W[idx]);
}

// ---------- generic bf16 WMMA GEMM: C = A[M,K] * Bt[N,K]^T (+bias) (+residual) ----------
template<bool BIAS, bool TRANS_OUT, bool OUT_BF16, bool RESID>
__global__ void gemm_wmma_kernel(const u16* __restrict__ A, const u16* __restrict__ Bt,
                                 const float* __restrict__ bias,
                                 const float* __restrict__ resid,
                                 void* __restrict__ outp, int M, int N, int K, int ldo) {
  int wid = threadIdx.x >> 5;
  int lane = threadIdx.x & 31;
  int wm = wid & 3, wn = wid >> 2;
  int m0 = blockIdx.y * 64 + wm * 16;
  int n0 = blockIdx.x * 128 + wn * 64;
  if (m0 >= M || n0 >= N) return;

  v8f acc[4] = {};
  for (int kk = 0; kk < K; kk += 32) {
    v16bf a = load_frag_a(A, K, m0, kk);
#pragma unroll
    for (int c = 0; c < 4; ++c) {
      int nc = n0 + 16 * c;
      if (nc < N) {
        v16bf b = load_frag_b(Bt, K, nc, kk);
        acc[c] = wmma_bf16(a, b, acc[c]);
      }
    }
  }
  int r = lane & 15, kh = lane >> 4;
#pragma unroll
  for (int c = 0; c < 4; ++c) {
    int nc = n0 + 16 * c;
    if (nc >= N) continue;
    int col = nc + r;
#pragma unroll
    for (int j = 0; j < 8; ++j) {
      int row = m0 + j + 8 * kh;
      float val = acc[c][j];
      if (BIAS)  val += bias[col];
      if (RESID) val += resid[(size_t)row * N + col];
      size_t o = TRANS_OUT ? (size_t)col * ldo + row : (size_t)row * ldo + col;
      if (OUT_BF16) ((u16*)outp)[o] = f2bf(val);
      else          ((float*)outp)[o] = val;
    }
  }
}

// ---------- fused flash attention with TDM-staged K/V tiles ----------
// grid (F/64, H), block 128 (4 waves). Wave w owns query rows f0+16w..+15.
// LDS: sQ[64][1024] + sK 2x[32][1024] + sV 2x[64][32] + sP 4x[16][32] = 268KB.
__global__ void flash_attn_kernel(const u16* __restrict__ qb,   // [F, D] bf16
                                  const u16* __restrict__ kb,   // [T, D] bf16
                                  const u16* __restrict__ vT,   // [D, T] bf16 (v transposed)
                                  const float* __restrict__ cbT,// [H, T] f32
                                  const float* __restrict__ mixing, // [H, D] f32
                                  u16* __restrict__ ctx) {      // [F, D] bf16
  extern __shared__ char smem_raw[];
  u16* sQ = (u16*)smem_raw;              // [64][1024]
  u16* sK = sQ + 64 * DM;                // 2 x [32][1024]
  u16* sV = sK + 2 * 32 * DM;            // 2 x [64][32]
  u16* sP = sV + 2 * 64 * 32;            // per-wave [16][32]

  const int h   = blockIdx.y;
  const int f0  = blockIdx.x * 64;
  const int tid = threadIdx.x;
  const int wid = tid >> 5;
  const int lane = tid & 31;
  const float scale = 0.125f;            // 1/sqrt(D/H) = 1/sqrt(64)

  const unsigned sK_off = (unsigned)(size_t)(void*)sK;   // flat low bits == LDS byte offset
  const unsigned sV_off = (unsigned)(size_t)(void*)sV;

#if HAS_TDM
  if (wid == 0) {   // prefetch tile 0 (overlaps with mixed-Q staging below)
    tdm_load_2d(sK_off, (u64)(size_t)kb, DM, 32, DM);
    tdm_load_2d(sV_off, (u64)(size_t)(vT + (size_t)h * DH * TN), 32, DH, TN);
  }
#endif

  // Stage mixed_q = q * mixing[h] into LDS as bf16.
  for (int e = tid * 4; e < 64 * DM; e += 128 * 4) {
    int row = e >> 10, d = e & (DM - 1);
    uint2  qv = *reinterpret_cast<const uint2*>(qb + (size_t)(f0 + row) * DM + d);
    float4 mx = *reinterpret_cast<const float4*>(mixing + (size_t)h * DM + d);
    unsigned lo = (unsigned)f2bf(bf2f((u16)(qv.x & 0xFFFF)) * mx.x) |
                  ((unsigned)f2bf(bf2f((u16)(qv.x >> 16))   * mx.y) << 16);
    unsigned hi = (unsigned)f2bf(bf2f((u16)(qv.y & 0xFFFF)) * mx.z) |
                  ((unsigned)f2bf(bf2f((u16)(qv.y >> 16))   * mx.w) << 16);
    uint2 ov; ov.x = lo; ov.y = hi;
    *reinterpret_cast<uint2*>(sQ + e) = ov;
  }
#if HAS_TDM
  wait_tensorcnt0();     // wave0 waits for tile 0; no-op elsewhere
#endif
  __syncthreads();

  const int r = lane & 15, kh = lane >> 4;
  u16* sPw = sP + wid * 16 * 32;

  float m_s[8], l_s[8];
#pragma unroll
  for (int j = 0; j < 8; ++j) { m_s[j] = -1e30f; l_s[j] = 0.f; }
  v8f ctxa[4] = {};

  for (int it = 0; it < TN / 32; ++it) {
    const int t0 = it * 32;
    int buf = it & 1;
#if HAS_TDM
    if (wid == 0 && it + 1 < TN / 32) {     // async prefetch of next tile
      int nb = buf ^ 1;
      tdm_load_2d(sK_off + (unsigned)nb * 32 * DM * 2,
                  (u64)(size_t)(kb + (size_t)(t0 + 32) * DM), DM, 32, DM);
      tdm_load_2d(sV_off + (unsigned)nb * 64 * 32 * 2,
                  (u64)(size_t)(vT + (size_t)h * DH * TN + (t0 + 32)), 32, DH, TN);
    }
#else
    __syncthreads();
    buf = 0;   // synchronous single-buffer fallback
    for (int e = tid * 8; e < 32 * DM; e += 128 * 8)
      *reinterpret_cast<uint4*>(sK + e) =
          *reinterpret_cast<const uint4*>(kb + (size_t)t0 * DM + e);
    for (int e = tid * 8; e < 64 * 32; e += 128 * 8) {
      int rr = e >> 5, cc = e & 31;
      *reinterpret_cast<uint4*>(sV + e) =
          *reinterpret_cast<const uint4*>(vT + (size_t)(h * DH + rr) * TN + t0 + cc);
    }
    __syncthreads();
#endif
    const u16* kbuf = sK + buf * 32 * DM;
    const u16* vbuf = sV + buf * 64 * 32;

    // S = mixedQ (16x1024) @ k_tile^T -> two 16x16 tiles (all frags from LDS)
    v8f S0 = {}, S1 = {};
    for (int kk = 0; kk < DM; kk += 32) {
      v16bf a  = load_frag_a(sQ, DM, wid * 16, kk);
      v16bf b0 = load_frag_b(kbuf, DM, 0, kk);
      v16bf b1 = load_frag_b(kbuf, DM, 16, kk);
      S0 = wmma_bf16(a, b0, S0);
      S1 = wmma_bf16(a, b1, S1);
    }
    float cb0 = cbT[(size_t)h * TN + t0 + r];
    float cb1 = cbT[(size_t)h * TN + t0 + 16 + r];
#pragma unroll
    for (int j = 0; j < 8; ++j) {
      S0[j] = (S0[j] + cb0) * scale;
      S1[j] = (S1[j] + cb1) * scale;
    }
    // Online softmax (row stats reduced across the 16 lanes of each half-wave).
#pragma unroll
    for (int j = 0; j < 8; ++j) {
      float sm = fmaxf(S0[j], S1[j]);
#pragma unroll
      for (int msk = 1; msk < 16; msk <<= 1) sm = fmaxf(sm, __shfl_xor(sm, msk, 32));
      float mn = fmaxf(m_s[j], sm);
      float alpha = __expf(m_s[j] - mn);
      m_s[j] = mn;
      float p0 = __expf(S0[j] - mn);
      float p1 = __expf(S1[j] - mn);
      float ps = p0 + p1;
#pragma unroll
      for (int msk = 1; msk < 16; msk <<= 1) ps += __shfl_xor(ps, msk, 32);
      l_s[j] = l_s[j] * alpha + ps;
#pragma unroll
      for (int c = 0; c < 4; ++c) ctxa[c][j] *= alpha;
      // Scatter P into per-wave LDS tile (row-major [16][32]) for A-frag re-load.
      sPw[(j + 8 * kh) * 32 + r]      = f2bf(p0);
      sPw[(j + 8 * kh) * 32 + 16 + r] = f2bf(p1);
    }
    asm volatile("s_wait_dscnt 0" ::: "memory");  // cross-lane LDS visibility (same wave)
    v16bf pf = load_frag_a(sPw, 32, 0, 0);
#pragma unroll
    for (int c = 0; c < 4; ++c) {
      v16bf bv_ = load_frag_b(vbuf, 32, c * 16, 0);
      ctxa[c] = wmma_bf16(pf, bv_, ctxa[c]);
    }
#if HAS_TDM
    wait_tensorcnt0();    // next tile landed (wave0); then release buffers
    __syncthreads();
#endif
  }
  // Normalize and store ctx as bf16 [F, D] (column block h*64..h*64+63).
#pragma unroll
  for (int c = 0; c < 4; ++c) {
#pragma unroll
    for (int j = 0; j < 8; ++j) {
      int row = f0 + wid * 16 + j + 8 * kh;
      int col = h * DH + c * 16 + r;
      ctx[(size_t)row * DM + col] = f2bf(ctxa[c][j] / l_s[j]);
    }
  }
}

// ---------- layernorm over rows of res [F, D] ----------
__global__ void layernorm_kernel(const float* __restrict__ res,
                                 const float* __restrict__ gamma,
                                 const float* __restrict__ beta,
                                 float* __restrict__ out) {
  __shared__ float red[2][8];
  int row = blockIdx.x, tid = threadIdx.x;
  const float* x = res + (size_t)row * DM;
  float s = 0.f, s2 = 0.f;
  for (int d = tid; d < DM; d += blockDim.x) { float v = x[d]; s += v; s2 += v * v; }
#pragma unroll
  for (int m = 1; m < 32; m <<= 1) { s += __shfl_xor(s, m, 32); s2 += __shfl_xor(s2, m, 32); }
  int wid = tid >> 5, lane = tid & 31;
  if (lane == 0) { red[0][wid] = s; red[1][wid] = s2; }
  __syncthreads();
  if (tid == 0) {
    float ts = 0.f, ts2 = 0.f;
    for (int w = 0; w < (int)(blockDim.x >> 5); ++w) { ts += red[0][w]; ts2 += red[1][w]; }
    red[0][0] = ts; red[1][0] = ts2;
  }
  __syncthreads();
  float mean = red[0][0] / DM;
  float var  = red[1][0] / DM - mean * mean;
  float inv  = rsqrtf(var + LNEPS);
  for (int d = tid; d < DM; d += blockDim.x)
    out[(size_t)row * DM + d] = (x[d] - mean) * inv * gamma[d] + beta[d];
}

// ---------- host driver ----------
extern "C" void kernel_launch(void* const* d_in, const int* in_sizes, int n_in,
                              void* d_out, int out_size, void* d_ws, size_t ws_size,
                              hipStream_t stream) {
  (void)in_sizes; (void)n_in; (void)out_size; (void)ws_size;
  const float* hidden = (const float*)d_in[0];
  const int*   fpos   = (const int*)d_in[1];
  const int*   tpos   = (const int*)d_in[2];
  const float* Wq     = (const float*)d_in[3];
  const float* Wk     = (const float*)d_in[4];
  const float* Wcb    = (const float*)d_in[5];
  const float* Wv     = (const float*)d_in[6];
  const float* bv     = (const float*)d_in[7];
  const float* mixing = (const float*)d_in[8];
  const float* Wd     = (const float*)d_in[9];
  const float* bd     = (const float*)d_in[10];
  const float* lng    = (const float*)d_in[11];
  const float* lnb    = (const float*)d_in[12];

  size_t off = 0;
  auto carve = [&](size_t bytes) {
    void* p = (char*)d_ws + off;
    off += (bytes + 255) & ~(size_t)255;
    return p;
  };
  u16*   from_bf = (u16*)carve((size_t)FN * DM * 2);
  u16*   to_bf   = (u16*)carve((size_t)TN * DM * 2);
  float* from_f  = (float*)carve((size_t)FN * DM * 4);
  u16*   WqT     = (u16*)carve((size_t)DM * DM * 2);
  u16*   WkT     = (u16*)carve((size_t)DM * DM * 2);
  u16*   WvT     = (u16*)carve((size_t)DM * DM * 2);
  u16*   WdT     = (u16*)carve((size_t)DM * DM * 2);
  u16*   WcbT    = (u16*)carve((size_t)HH * DM * 2);
  u16*   q_bf    = (u16*)carve((size_t)FN * DM * 2);
  u16*   k_bf    = (u16*)carve((size_t)TN * DM * 2);
  u16*   vT_bf   = (u16*)carve((size_t)DM * TN * 2);
  float* cbT     = (float*)carve((size_t)HH * TN * 4);
  u16*   ctx_bf  = (u16*)carve((size_t)FN * DM * 2);
  float* res_f   = (float*)carve((size_t)FN * DM * 4);

  // 1) gather + convert
  {
    int total = FN * DM;
    dim3 g((total + 255) / 256), b(256);
    gather_convert_kernel<<<g, b, 0, stream>>>(hidden, fpos, from_bf, from_f, FN);
    gather_convert_kernel<<<g, b, 0, stream>>>(hidden, tpos, to_bf, nullptr, TN);
  }
  // 2) weight transpose/convert
  {
    dim3 b(256);
    dim3 gw((DM * DM + 255) / 256);
    transpose_convert_kernel<<<gw, b, 0, stream>>>(Wq, WqT, DM, DM);
    transpose_convert_kernel<<<gw, b, 0, stream>>>(Wk, WkT, DM, DM);
    transpose_convert_kernel<<<gw, b, 0, stream>>>(Wv, WvT, DM, DM);
    transpose_convert_kernel<<<gw, b, 0, stream>>>(Wd, WdT, DM, DM);
    dim3 gc((DM * HH + 255) / 256);
    transpose_convert_kernel<<<gc, b, 0, stream>>>(Wcb, WcbT, DM, HH);
  }
  // 3) projections (WMMA GEMMs)
  {
    dim3 b(256);
    dim3 g((DM + 127) / 128, FN / 64);
    gemm_wmma_kernel<false, false, true, false><<<g, b, 0, stream>>>(
        from_bf, WqT, nullptr, nullptr, q_bf, FN, DM, DM, DM);
    gemm_wmma_kernel<false, false, true, false><<<g, b, 0, stream>>>(
        to_bf, WkT, nullptr, nullptr, k_bf, TN, DM, DM, DM);
    gemm_wmma_kernel<true, true, true, false><<<g, b, 0, stream>>>(
        to_bf, WvT, bv, nullptr, vT_bf, TN, DM, DM, /*ldo=*/TN);
    dim3 gcb(1, TN / 64);
    gemm_wmma_kernel<false, true, false, false><<<gcb, b, 0, stream>>>(
        to_bf, WcbT, nullptr, nullptr, cbT, TN, HH, DM, /*ldo=*/TN);
  }
  // 4) fused attention (flash, online softmax, TDM double-buffered K/V tiles)
  {
    size_t smem = (size_t)64 * DM * 2      // sQ
                + (size_t)2 * 32 * DM * 2  // sK double buffer
                + (size_t)2 * 64 * 32 * 2  // sV double buffer
                + (size_t)4 * 16 * 32 * 2; // sP
    dim3 g(FN / 64, HH), b(128);
    flash_attn_kernel<<<g, b, smem, stream>>>(q_bf, k_bf, vT_bf, cbT, mixing, ctx_bf);
  }
  // 5) output projection + bias + residual
  {
    dim3 b(256);
    dim3 g((DM + 127) / 128, FN / 64);
    gemm_wmma_kernel<true, false, false, true><<<g, b, 0, stream>>>(
        ctx_bf, WdT, bd, from_f, res_f, FN, DM, DM, DM);
  }
  // 6) layernorm -> d_out
  {
    dim3 g(FN), b(256);
    layernorm_kernel<<<g, b, 0, stream>>>(res_f, lng, lnb, (float*)d_out);
  }
}
